// CustomMixtralSparseMoeBlock_8108898254884
// MI455X (gfx1250) — matrix operand order repaired and checked by reading, not compiled, not run on hardware
//
#include <hip/hip_runtime.h>
#include <hip/hip_bf16.h>
#include <math.h>

// Problem constants (match reference)
#define TT   8192      // T = B*S tokens
#define HH   1024      // hidden
#define FF   3584      // ffn
#define EE   8         // real experts
#define NG   10        // E + N router logits

typedef __attribute__((ext_vector_type(16))) __bf16 v16bf;
typedef __attribute__((ext_vector_type(8)))  __bf16 v8bf;
typedef __attribute__((ext_vector_type(8)))  float  v8f;

// fragment strides in the swizzled B layout (elements)
#define ST1J  ((HH >> 5) * 512)   // stage1: j -> j+1 (one 16-row tile over H)
#define ST2J  ((FF >> 5) * 512)   // stage2: j -> j+1 (one 16-row tile over F)

__device__ __forceinline__ v8f zero8() {
    v8f z;
#pragma unroll
    for (int i = 0; i < 8; ++i) z[i] = 0.0f;
    return z;
}

__device__ __forceinline__ v16bf load_a_frag(const __bf16* row, int k0, int hl) {
    v8bf lo = *(const v8bf*)(row + k0 + hl * 16);
    v8bf hi = *(const v8bf*)(row + k0 + hl * 16 + 8);
    v16bf a;
#pragma unroll
    for (int i = 0; i < 8; ++i) { a[i] = lo[i]; a[8 + i] = hi[i]; }
    return a;
}

// ---------------------------------------------------------------------------
// init: zero the dense output region and the per-expert counters
// ---------------------------------------------------------------------------
__global__ __launch_bounds__(256) void moe_init(float4* __restrict__ out4,
                                                int* __restrict__ counts,
                                                int n4) {
    int i = blockIdx.x * blockDim.x + threadIdx.x;
    if (i < n4) out4[i] = make_float4(0.f, 0.f, 0.f, 0.f);
    if (i < EE) counts[i] = 0;
}

// ---------------------------------------------------------------------------
// x -> bf16 row-major
// ---------------------------------------------------------------------------
__global__ __launch_bounds__(256) void moe_cvt_x(const float* __restrict__ x,
                                                 __bf16* __restrict__ xbf,
                                                 int n8) {
    int i = blockIdx.x * blockDim.x + threadIdx.x;
    if (i >= n8) return;
    const float4* s = (const float4*)(x + (size_t)i * 8);
    float4 a = s[0], b = s[1];
    v8bf o;
    o[0]=(__bf16)a.x; o[1]=(__bf16)a.y; o[2]=(__bf16)a.z; o[3]=(__bf16)a.w;
    o[4]=(__bf16)b.x; o[5]=(__bf16)b.y; o[6]=(__bf16)b.z; o[7]=(__bf16)b.w;
    *(v8bf*)(xbf + (size_t)i * 8) = o;
}

// ---------------------------------------------------------------------------
// Per-expert weight conversion: f32 [R][C] -> bf16 in WMMA B-fragment layout.
// Tile = 16 rows (N) x 32 cols (K); lane = ((col>>4)&1)*16 | (row&15),
// elems 0..15 = 16 contiguous cols.  Fragment base =
// ((rt * (C/32) + ct) * 32 + lane) * 16   (each tile 1KB, lane-contiguous)
// blockIdx.z: 0 -> w1[e], 1 -> w3[e], 2 -> w2[e]
// ---------------------------------------------------------------------------
__global__ __launch_bounds__(256) void moe_cvt_w(
    const float* __restrict__ w1, const float* __restrict__ w3,
    const float* __restrict__ w2, __bf16* __restrict__ wb1,
    __bf16* __restrict__ wb3, __bf16* __restrict__ wb2, int e)
{
    const float* src; __bf16* dst; int R, C;
    if (blockIdx.z == 0)      { src = w1 + (size_t)e * FF * HH; dst = wb1; R = FF; C = HH; }
    else if (blockIdx.z == 1) { src = w3 + (size_t)e * FF * HH; dst = wb3; R = FF; C = HH; }
    else                      { src = w2 + (size_t)e * HH * FF; dst = wb2; R = HH; C = FF; }

    int frag = blockIdx.x * blockDim.x + threadIdx.x;   // one 16-col chunk
    int nchunk = C >> 4;
    if (frag >= R * nchunk) return;
    int row = frag / nchunk, c16 = frag % nchunk;

    const float* s = src + (size_t)row * C + c16 * 16;
    float4 u0 = ((const float4*)s)[0];
    float4 u1 = ((const float4*)s)[1];
    float4 u2 = ((const float4*)s)[2];
    float4 u3 = ((const float4*)s)[3];
    v16bf o;
    o[0]=(__bf16)u0.x;  o[1]=(__bf16)u0.y;  o[2]=(__bf16)u0.z;  o[3]=(__bf16)u0.w;
    o[4]=(__bf16)u1.x;  o[5]=(__bf16)u1.y;  o[6]=(__bf16)u1.z;  o[7]=(__bf16)u1.w;
    o[8]=(__bf16)u2.x;  o[9]=(__bf16)u2.y;  o[10]=(__bf16)u2.z; o[11]=(__bf16)u2.w;
    o[12]=(__bf16)u3.x; o[13]=(__bf16)u3.y; o[14]=(__bf16)u3.z; o[15]=(__bf16)u3.w;

    int rt = row >> 4, ct = c16 >> 1;
    int lane = ((c16 & 1) << 4) | (row & 15);
    __bf16* d = dst + ((((size_t)rt * (C >> 5) + ct) * 32 + lane) << 4);
    *(v16bf*)d = o;
}

// ---------------------------------------------------------------------------
// router: logits, softmax, top-2, combine weights, per-expert gather lists
// ---------------------------------------------------------------------------
__global__ __launch_bounds__(256) void moe_router(
    const float* __restrict__ x, const float* __restrict__ gate_w,
    const float* __restrict__ gate2_w, float* __restrict__ rlogits,
    int* __restrict__ counts, int* __restrict__ idxbuf,
    float* __restrict__ wtbuf)
{
    __shared__ float gsh[NG * HH];  // 40 KB
    for (int i = threadIdx.x; i < NG * HH; i += blockDim.x)
        gsh[i] = (i < EE * HH) ? gate_w[i] : gate2_w[i - EE * HH];
    __syncthreads();

    const int lane = threadIdx.x & 31;
    const int wave = threadIdx.x >> 5;
    const int tok  = blockIdx.x * 8 + wave;
    if (tok >= TT) return;

    const float* xr = x + (size_t)tok * HH;
    float acc[NG];
#pragma unroll
    for (int g = 0; g < NG; ++g) acc[g] = 0.f;

    for (int k = lane; k < HH; k += 32) {
        float xv = xr[k];
#pragma unroll
        for (int g = 0; g < NG; ++g) acc[g] = fmaf(xv, gsh[g * HH + k], acc[g]);
    }
#pragma unroll
    for (int off = 16; off > 0; off >>= 1) {
#pragma unroll
        for (int g = 0; g < NG; ++g) acc[g] += __shfl_xor(acc[g], off, 32);
    }

    if (lane == 0) {
        float mx = acc[0];
#pragma unroll
        for (int g = 1; g < NG; ++g) mx = fmaxf(mx, acc[g]);
        float rw[NG], sum = 0.f;
#pragma unroll
        for (int g = 0; g < NG; ++g) { rw[g] = __expf(acc[g] - mx); sum += rw[g]; }
        float inv = 1.f / sum;
#pragma unroll
        for (int g = 0; g < NG; ++g) {
            rw[g] *= inv;
            rlogits[(size_t)tok * NG + g] = acc[g];
        }
        int k1 = 0;
#pragma unroll
        for (int g = 1; g < NG; ++g) if (rw[g] > rw[k1]) k1 = g;
        int k2 = (k1 == 0) ? 1 : 0;
#pragma unroll
        for (int g = 0; g < NG; ++g) if (g != k2 && g != k1 && rw[g] > rw[k2]) k2 = g;

        float tw1 = rw[k1], tw2 = rw[k2];
        bool  m1 = (k1 < EE), m2 = (k2 < EE);
        float sw1 = m1 ? tw1 : 0.f, sw2 = m2 ? tw2 : 0.f;
        float denom = sw1 + sw2;
        if (denom == 0.f) denom = 1.f;
        float wg1 = sw1 / denom + (m1 ? 0.f : tw1);
        float wg2 = sw2 / denom + (m2 ? 0.f : tw2);

        if (m1) {
            int p = atomicAdd(&counts[k1], 1);
            idxbuf[k1 * TT + p] = tok; wtbuf[k1 * TT + p] = wg1;
        }
        if (m2) {
            int p = atomicAdd(&counts[k2], 1);
            idxbuf[k2 * TT + p] = tok; wtbuf[k2 * TT + p] = wg2;
        }
    }
}

// ---------------------------------------------------------------------------
// stage1: hgate[r, f] = silu(x_g . w1e^T) * (x_g . w3e^T)
// Software-pipelined: double-buffered A + B fragments so WMMAs issue while
// the next k-step's loads are in flight (partial s_wait_loadcnt, not 0).
// ---------------------------------------------------------------------------
__global__ __launch_bounds__(256) void moe_stage1(
    const __bf16* __restrict__ xbf,
    const __bf16* __restrict__ wb1, const __bf16* __restrict__ wb3,
    const int* __restrict__ counts, const int* __restrict__ idxbuf,
    __bf16* __restrict__ hgate, int e)
{
    const int cnt   = counts[e];
    const int mtile = blockIdx.y * 32;
    if (mtile >= cnt) return;

    const int lane = threadIdx.x & 31;
    const int wave = threadIdx.x >> 5;
    const int mw = wave & 1, nw = wave >> 1;
    const int mbase = mtile + mw * 16;
    const int nbase = blockIdx.x * 256 + nw * 64;
    const int hl  = lane >> 4;
    const int l16 = lane & 15;

    int arow = mbase + l16;
    int ar   = arow < cnt ? arow : cnt - 1;
    const int tok = idxbuf[e * TT + ar];
    const __bf16* xr = xbf + (size_t)tok * HH;

    const __bf16* pB1 = wb1 + (size_t)(nbase >> 4) * ST1J + lane * 16;
    const __bf16* pB3 = wb3 + (size_t)(nbase >> 4) * ST1J + lane * 16;

    v8f acc1[4], acc3[4];
#pragma unroll
    for (int j = 0; j < 4; ++j) { acc1[j] = zero8(); acc3[j] = zero8(); }

    v16bf A0, A1, B1a[4], B3a[4], B1b[4], B3b[4];

    // prologue: fill buffer 0 with k = 0
    A0 = load_a_frag(xr, 0, hl);
#pragma unroll
    for (int j = 0; j < 4; ++j) {
        B1a[j] = *(const v16bf*)(pB1 + (size_t)j * ST1J);
        B3a[j] = *(const v16bf*)(pB3 + (size_t)j * ST1J);
    }

    for (int k0 = 0; k0 < HH; k0 += 64) {
        const size_t ko1 = (size_t)((k0 + 32) >> 5) * 512;
        const size_t ko2 = (size_t)((k0 + 64) >> 5) * 512;  // overreads once; pad

        // prefetch k0+32 into buffer 1
        A1 = load_a_frag(xr, k0 + 32, hl);
#pragma unroll
        for (int j = 0; j < 4; ++j) {
            B1b[j] = *(const v16bf*)(pB1 + (size_t)j * ST1J + ko1);
            B3b[j] = *(const v16bf*)(pB3 + (size_t)j * ST1J + ko1);
        }
        // compute buffer 0 (k0)
#pragma unroll
        for (int j = 0; j < 4; ++j) {
            acc1[j] = __builtin_amdgcn_wmma_f32_16x16x32_bf16(
                false, A0, false, B1a[j], (short)0, acc1[j], false, false);
            acc3[j] = __builtin_amdgcn_wmma_f32_16x16x32_bf16(
                false, A0, false, B3a[j], (short)0, acc3[j], false, false);
        }
        // prefetch k0+64 into buffer 0 (dead on last iter; lands in ws pad)
        A0 = load_a_frag(xr, k0 + 64, hl);
#pragma unroll
        for (int j = 0; j < 4; ++j) {
            B1a[j] = *(const v16bf*)(pB1 + (size_t)j * ST1J + ko2);
            B3a[j] = *(const v16bf*)(pB3 + (size_t)j * ST1J + ko2);
        }
        // compute buffer 1 (k0+32)
#pragma unroll
        for (int j = 0; j < 4; ++j) {
            acc1[j] = __builtin_amdgcn_wmma_f32_16x16x32_bf16(
                false, A1, false, B1b[j], (short)0, acc1[j], false, false);
            acc3[j] = __builtin_amdgcn_wmma_f32_16x16x32_bf16(
                false, A1, false, B3b[j], (short)0, acc3[j], false, false);
        }
    }

    // epilogue: silu(acc1)*acc3 -> bf16 hgate.  C layout: M = v + 8*hl, N = l16
#pragma unroll
    for (int j = 0; j < 4; ++j) {
#pragma unroll
        for (int v = 0; v < 8; ++v) {
            int row = mbase + v + 8 * hl;
            if (row < cnt) {
                float g = acc1[j][v];
                float s = g / (1.f + __expf(-g));
                float val = s * acc3[j][v];
                hgate[(size_t)row * FF + nbase + j * 16 + l16] = (__bf16)val;
            }
        }
    }
}

// ---------------------------------------------------------------------------
// stage2: out[token, h] += wt[r] * (hgate[r,:] . w2e[h,:])
// Same double-buffered pipeline.  Non-atomic accumulate is race-free
// (per-expert kernels serialize on the stream; token unique per list).
// ---------------------------------------------------------------------------
__global__ __launch_bounds__(256) void moe_stage2(
    const __bf16* __restrict__ hgate, const __bf16* __restrict__ wb2,
    const int* __restrict__ counts, const int* __restrict__ idxbuf,
    const float* __restrict__ wtbuf, float* __restrict__ out, int e)
{
    const int cnt   = counts[e];
    const int mtile = blockIdx.y * 32;
    if (mtile >= cnt) return;

    const int lane = threadIdx.x & 31;
    const int wave = threadIdx.x >> 5;
    const int mw = wave & 1, nw = wave >> 1;
    const int mbase = mtile + mw * 16;
    const int nbase = blockIdx.x * 256 + nw * 64;   // H columns
    const int hl  = lane >> 4;
    const int l16 = lane & 15;

    int arow = mbase + l16;
    int ar   = arow < cnt ? arow : cnt - 1;
    const __bf16* hr = hgate + (size_t)ar * FF;

    const __bf16* pB = wb2 + (size_t)(nbase >> 4) * ST2J + lane * 16;

    v8f acc[4];
#pragma unroll
    for (int j = 0; j < 4; ++j) acc[j] = zero8();

    v16bf A0, A1, Ba[4], Bb[4];

    A0 = load_a_frag(hr, 0, hl);
#pragma unroll
    for (int j = 0; j < 4; ++j)
        Ba[j] = *(const v16bf*)(pB + (size_t)j * ST2J);

    for (int k0 = 0; k0 < FF; k0 += 64) {
        const size_t ko1 = (size_t)((k0 + 32) >> 5) * 512;
        const size_t ko2 = (size_t)((k0 + 64) >> 5) * 512;

        A1 = load_a_frag(hr, k0 + 32, hl);
#pragma unroll
        for (int j = 0; j < 4; ++j)
            Bb[j] = *(const v16bf*)(pB + (size_t)j * ST2J + ko1);
#pragma unroll
        for (int j = 0; j < 4; ++j)
            acc[j] = __builtin_amdgcn_wmma_f32_16x16x32_bf16(
                false, A0, false, Ba[j], (short)0, acc[j], false, false);

        A0 = load_a_frag(hr, k0 + 64, hl);
#pragma unroll
        for (int j = 0; j < 4; ++j)
            Ba[j] = *(const v16bf*)(pB + (size_t)j * ST2J + ko2);
#pragma unroll
        for (int j = 0; j < 4; ++j)
            acc[j] = __builtin_amdgcn_wmma_f32_16x16x32_bf16(
                false, A1, false, Bb[j], (short)0, acc[j], false, false);
    }

#pragma unroll
    for (int j = 0; j < 4; ++j) {
#pragma unroll
        for (int v = 0; v < 8; ++v) {
            int row = mbase + v + 8 * hl;
            if (row < cnt) {
                int   tok = idxbuf[e * TT + row];
                float w   = wtbuf[e * TT + row];
                size_t o  = (size_t)tok * HH + nbase + j * 16 + l16;
                out[o] += w * acc[j][v];
            }
        }
    }
}

// ---------------------------------------------------------------------------
// launcher
// ---------------------------------------------------------------------------
extern "C" void kernel_launch(void* const* d_in, const int* in_sizes, int n_in,
                              void* d_out, int out_size, void* d_ws, size_t ws_size,
                              hipStream_t stream) {
    (void)in_sizes; (void)n_in; (void)out_size; (void)ws_size;

    const float* x       = (const float*)d_in[0];
    const float* gate_w  = (const float*)d_in[1];
    const float* gate2_w = (const float*)d_in[2];
    const float* w1      = (const float*)d_in[3];
    const float* w2      = (const float*)d_in[4];
    const float* w3      = (const float*)d_in[5];

    float* out     = (float*)d_out;
    float* rlogits = out + (size_t)TT * HH;

    // workspace layout (~90 MB + pads for pipeline overreads)
    char* ws = (char*)d_ws;
    size_t off = 0;
    int*    counts = (int*)(ws + off);    off += 256;
    int*    idxbuf = (int*)(ws + off);    off += (size_t)EE * TT * 4;
    float*  wtbuf  = (float*)(ws + off);  off += (size_t)EE * TT * 4;
    __bf16* xbf    = (__bf16*)(ws + off); off += (size_t)TT * HH * 2;
    __bf16* hgate  = (__bf16*)(ws + off); off += (size_t)TT * FF * 2;
    __bf16* wb1    = (__bf16*)(ws + off); off += (size_t)FF * HH * 2;
    __bf16* wb3    = (__bf16*)(ws + off); off += (size_t)FF * HH * 2;
    __bf16* wb2    = (__bf16*)(ws + off); off += (size_t)HH * FF * 2;
    off += 4096;  // pad: absorbs dead prefetch overreads from wb2

    {
        int n4 = (TT * HH) / 4;
        moe_init<<<(n4 + 255) / 256, 256, 0, stream>>>((float4*)out, counts, n4);
    }
    {
        int n8 = (TT * HH) / 8;
        moe_cvt_x<<<(n8 + 255) / 256, 256, 0, stream>>>(x, xbf, n8);
    }
    moe_router<<<TT / 8, 256, 0, stream>>>(x, gate_w, gate2_w, rlogits,
                                           counts, idxbuf, wtbuf);
    {
        int nfrag = (FF * HH) / 16;
        dim3 cgrid((nfrag + 255) / 256, 1, 3);
        for (int e = 0; e < EE; ++e) {
            moe_cvt_w<<<cgrid, 256, 0, stream>>>(w1, w3, w2, wb1, wb3, wb2, e);
            moe_stage1<<<dim3(FF / 256, TT / 32), 256, 0, stream>>>(
                xbf, wb1, wb3, counts, idxbuf, hgate, e);
            moe_stage2<<<dim3(HH / 256, TT / 32), 256, 0, stream>>>(
                hgate, wb2, counts, idxbuf, wtbuf, out, e);
        }
    }
}
